// GraphAttentionNetwork_79637283603301
// MI455X (gfx1250) — matrix operand order repaired and checked by reading, not compiled, not run on hardware
//
#include <hip/hip_runtime.h>
#include <hip/hip_bf16.h>
#include <math.h>

typedef __attribute__((ext_vector_type(16))) _Float16 v16h;
typedef __attribute__((ext_vector_type(8)))  _Float16 v8h;
typedef __attribute__((ext_vector_type(4)))  _Float16 v4h;
typedef __attribute__((ext_vector_type(8)))  float    v8f;

#define N_NODES 50000
#define N_EDGES 800000
#define E_TOT   850000   // edges + self loops
#define DIM     128
#define NEG_SLOPE 0.2f

// ---------------------------------------------------------------------------
// x (f32) -> f16
// ---------------------------------------------------------------------------
__global__ void convert_f16_kernel(const float* __restrict__ in,
                                   _Float16* __restrict__ out) {
  const int i = blockIdx.x * blockDim.x + threadIdx.x;
  if (i < N_NODES * DIM) out[i] = (_Float16)in[i];
}

// W[k][n] (f32, row-major) -> Wt[n][k] (f16, K-contiguous)
__global__ void transpose_w_kernel(const float* __restrict__ W,
                                   _Float16* __restrict__ Wt) {
  const int t = blockIdx.x * blockDim.x + threadIdx.x;  // t < 128*128
  const int n = t >> 7, k = t & 127;
  Wt[t] = (_Float16)W[k * DIM + n];
}

// ---------------------------------------------------------------------------
// GEMM: Hout[N,128] (f16) = Xh[N,128] (f16) @ W (via Wt[n][k] f16).
// One block = 16 nodes; 8 waves, wave w owns cols [16w,16w+16).
// All fragments are K-contiguous -> 16B vector loads, no per-step converts.
// ---------------------------------------------------------------------------
__global__ __launch_bounds__(256) void gemm128_kernel(
    const _Float16* __restrict__ Xh, const _Float16* __restrict__ Wt,
    _Float16* __restrict__ Hout) {
  __shared__ _Float16 lA[16 * DIM];              // 4 KB A tile (f16)
  const int node0 = blockIdx.x * 16;
  const int t = threadIdx.x;

  // cooperative A-tile load: 4 KB = 256 x 16B, one uint4 per thread
  ((uint4*)lA)[t] = ((const uint4*)(Xh + (size_t)node0 * DIM))[t];
  __syncthreads();

  const int lane = t & 31;
  const int wave = t >> 5;
  const int m    = lane & 15;
  const bool hi  = lane >= 16;
  const int n    = wave * 16 + m;
  const int ka   = hi ? 16 : 0;                  // A: K-split in blocks of 16
  const int kb   = hi ? 16 : 0;                  // B: K-split in blocks of 16

  const _Float16* arow = lA + m * DIM;
  const _Float16* wcol = Wt + (size_t)n * DIM;

  v8f acc = {};
#pragma unroll
  for (int k0 = 0; k0 < DIM; k0 += 32) {
    v8h a0 = *(const v8h*)(arow + k0 + ka);        // K = k0+ka .. +7
    v8h a1 = *(const v8h*)(arow + k0 + ka + 8);    // K = k0+ka+8 .. +15
    v16h a = __builtin_shufflevector(a0, a1, 0, 1, 2, 3, 4, 5, 6, 7,
                                     8, 9, 10, 11, 12, 13, 14, 15);
    v8h b0 = *(const v8h*)(wcol + k0 + kb);        // K = k0+kb .. +7
    v8h b1 = *(const v8h*)(wcol + k0 + kb + 8);    // K = k0+kb+8 .. +15
    v16h b = __builtin_shufflevector(b0, b1, 0, 1, 2, 3, 4, 5, 6, 7,
                                     8, 9, 10, 11, 12, 13, 14, 15);
    acc = __builtin_amdgcn_wmma_f32_16x16x32_f16(false, a, false, b,
                                                 (short)0, acc, false, false);
  }

#pragma unroll
  for (int r = 0; r < 8; ++r) {
    Hout[(size_t)(node0 + r + (hi ? 8 : 0)) * DIM + n] = (_Float16)acc[r];
  }
}

// ---------------------------------------------------------------------------
// Per-node prep: logits al_s/al_d, init m=-inf, z=0, zero acc row.
// One wave per node. H==4: chunk j (cols 32j..32j+31) is head j; H==1 flat.
// ---------------------------------------------------------------------------
__global__ __launch_bounds__(256) void node_prep_kernel(
    const _Float16* __restrict__ Hf, const float* __restrict__ a_src,
    const float* __restrict__ a_dst, float* __restrict__ als,
    float* __restrict__ ald, float* __restrict__ mbuf, float* __restrict__ zbuf,
    float* __restrict__ acc, int H) {
  const int node = (blockIdx.x * blockDim.x + threadIdx.x) >> 5;
  const int lane = threadIdx.x & 31;
  if (node >= N_NODES) return;

  const _Float16* hrow = Hf + (size_t)node * DIM;
  float s[4], d[4];
#pragma unroll
  for (int j = 0; j < 4; ++j) {
    const float hv = (float)hrow[j * 32 + lane];
    s[j] = hv * a_src[j * 32 + lane];   // [4,32] and [1,128] index identically
    d[j] = hv * a_dst[j * 32 + lane];
    acc[(size_t)node * DIM + j * 32 + lane] = 0.0f;
  }
#pragma unroll
  for (int j = 0; j < 4; ++j) {
    for (int off = 16; off > 0; off >>= 1) {
      s[j] += __shfl_xor(s[j], off, 32);
      d[j] += __shfl_xor(d[j], off, 32);
    }
  }
  if (lane == 0) {
    if (H == 4) {
#pragma unroll
      for (int j = 0; j < 4; ++j) {
        als[node * 4 + j] = s[j];
        ald[node * 4 + j] = d[j];
        mbuf[node * 4 + j] = -INFINITY;
        zbuf[node * 4 + j] = 0.0f;
      }
    } else {
      als[node] = s[0] + s[1] + s[2] + s[3];
      ald[node] = d[0] + d[1] + d[2] + d[3];
      mbuf[node] = -INFINITY;
      zbuf[node] = 0.0f;
    }
  }
}

// float atomic max via sign-split int/uint atomics (works with -inf init)
__device__ __forceinline__ void atomicMaxF32(float* addr, float val) {
  if (val >= 0.0f) atomicMax((int*)addr, __float_as_int(val));
  else             atomicMin((unsigned int*)addr, __float_as_uint(val));
}

__device__ __forceinline__ float leaky(float v) {
  return v > 0.0f ? v : NEG_SLOPE * v;
}

// pass 1: e = leaky(al_s[src]+al_d[dst]); cache it; segment max over dst
template <int H>
__global__ void edge_logit_max_kernel(const int* __restrict__ srcI,
                                      const int* __restrict__ dstI,
                                      const float* __restrict__ als,
                                      const float* __restrict__ ald,
                                      float* __restrict__ ebuf,
                                      float* __restrict__ mbuf) {
  const int e = blockIdx.x * blockDim.x + threadIdx.x;
  if (e >= E_TOT) return;
  const int s = (e < N_EDGES) ? srcI[e] : (e - N_EDGES);
  const int d = (e < N_EDGES) ? dstI[e] : (e - N_EDGES);
#pragma unroll
  for (int h = 0; h < H; ++h) {
    const float ev = leaky(als[s * H + h] + ald[d * H + h]);
    ebuf[(size_t)e * H + h] = ev;
    atomicMaxF32(&mbuf[d * H + h], ev);
  }
}

// pass 2: z[dst] += exp(e - m[dst])   (no src gather needed)
template <int H>
__global__ void edge_sum_kernel(const int* __restrict__ dstI,
                                const float* __restrict__ ebuf,
                                const float* __restrict__ mbuf,
                                float* __restrict__ zbuf) {
  const int e = blockIdx.x * blockDim.x + threadIdx.x;
  if (e >= E_TOT) return;
  const int d = (e < N_EDGES) ? dstI[e] : (e - N_EDGES);
#pragma unroll
  for (int h = 0; h < H; ++h) {
    atomicAdd(&zbuf[d * H + h],
              expf(ebuf[(size_t)e * H + h] - mbuf[d * H + h]));
  }
}

// pass 3: acc[dst] += alpha * h[src]; wave per edge, lane -> 4 cols (f16 gather)
template <int H>
__global__ __launch_bounds__(256) void edge_aggr_kernel(
    const int* __restrict__ srcI, const int* __restrict__ dstI,
    const float* __restrict__ ebuf, const float* __restrict__ mbuf,
    const float* __restrict__ zbuf, const _Float16* __restrict__ Hf,
    float* __restrict__ acc) {
  const int e    = (blockIdx.x * blockDim.x + threadIdx.x) >> 5;
  const int lane = threadIdx.x & 31;
  if (e >= E_TOT) return;
  const int s = (e < N_EDGES) ? srcI[e] : (e - N_EDGES);
  const int d = (e < N_EDGES) ? dstI[e] : (e - N_EDGES);

  const int h = (H == 4) ? (lane >> 3) : 0;   // cols 4*lane..4*lane+3
  const float ah = expf(ebuf[(size_t)e * H + h] - mbuf[d * H + h]) /
                   (zbuf[d * H + h] + 1e-16f);

  const int c = lane * 4;
  const v4h hv = *(const v4h*)(Hf + (size_t)s * DIM + c);
  float* ap = acc + (size_t)d * DIM + c;
  atomicAdd(ap + 0, ah * (float)hv[0]);
  atomicAdd(ap + 1, ah * (float)hv[1]);
  atomicAdd(ap + 2, ah * (float)hv[2]);
  atomicAdd(ap + 3, ah * (float)hv[3]);
}

// next-layer input = relu(acc + bias), stored f16
__global__ void relu_bias_kernel(const float* __restrict__ acc,
                                 const float* __restrict__ bias,
                                 _Float16* __restrict__ outp) {
  const int i = blockIdx.x * blockDim.x + threadIdx.x;
  if (i >= N_NODES * DIM) return;
  const float v = acc[i] + bias[i & (DIM - 1)];
  outp[i] = (_Float16)(v > 0.0f ? v : 0.0f);
}

// final: h = acc + b2; LayerNorm over DIM; one wave per node
__global__ __launch_bounds__(256) void ln_kernel(const float* __restrict__ acc,
                                                 const float* __restrict__ bias,
                                                 const float* __restrict__ gamma,
                                                 const float* __restrict__ beta,
                                                 float* __restrict__ out) {
  const int node = (blockIdx.x * blockDim.x + threadIdx.x) >> 5;
  const int lane = threadIdx.x & 31;
  if (node >= N_NODES) return;

  float v[4];
  float s = 0.0f;
#pragma unroll
  for (int j = 0; j < 4; ++j) {
    v[j] = acc[(size_t)node * DIM + j * 32 + lane] + bias[j * 32 + lane];
    s += v[j];
  }
  for (int off = 16; off > 0; off >>= 1) s += __shfl_xor(s, off, 32);
  const float mu = s * (1.0f / DIM);
  float var = 0.0f;
#pragma unroll
  for (int j = 0; j < 4; ++j) { const float t2 = v[j] - mu; var += t2 * t2; }
  for (int off = 16; off > 0; off >>= 1) var += __shfl_xor(var, off, 32);
  const float inv = rsqrtf(var * (1.0f / DIM) + 1e-5f);
#pragma unroll
  for (int j = 0; j < 4; ++j) {
    out[(size_t)node * DIM + j * 32 + lane] =
        (v[j] - mu) * inv * gamma[j * 32 + lane] + beta[j * 32 + lane];
  }
}

// ---------------------------------------------------------------------------
extern "C" void kernel_launch(void* const* d_in, const int* in_sizes, int n_in,
                              void* d_out, int out_size, void* d_ws, size_t ws_size,
                              hipStream_t stream) {
  const float* x    = (const float*)d_in[0];
  const int*   ei   = (const int*)d_in[1];
  const float* Wl[3]  = {(const float*)d_in[2], (const float*)d_in[6],
                         (const float*)d_in[10]};
  const float* asl[3] = {(const float*)d_in[3], (const float*)d_in[7],
                         (const float*)d_in[11]};
  const float* adl[3] = {(const float*)d_in[4], (const float*)d_in[8],
                         (const float*)d_in[12]};
  const float* bl[3]  = {(const float*)d_in[5], (const float*)d_in[9],
                         (const float*)d_in[13]};
  const float* gam  = (const float*)d_in[14];
  const float* bet  = (const float*)d_in[15];

  const int* srcI = ei;             // edge_index[0]
  const int* dstI = ei + N_EDGES;   // edge_index[1]

  char* ws = (char*)d_ws;
  size_t off = 0;
  auto alloc = [&](size_t bytes) {
    void* p = ws + off;
    off += (bytes + 255) & ~(size_t)255;
    return p;
  };
  _Float16* inbuf = (_Float16*)alloc((size_t)N_NODES * DIM * 2); // f16 gemm input
  _Float16* hfeat = (_Float16*)alloc((size_t)N_NODES * DIM * 2); // f16 features
  _Float16* Wt    = (_Float16*)alloc((size_t)DIM * DIM * 2);     // f16 W^T
  float* acc  = (float*)alloc((size_t)N_NODES * DIM * 4);
  float* als  = (float*)alloc((size_t)N_NODES * 4 * 4);
  float* ald  = (float*)alloc((size_t)N_NODES * 4 * 4);
  float* mb   = (float*)alloc((size_t)N_NODES * 4 * 4);
  float* zb   = (float*)alloc((size_t)N_NODES * 4 * 4);
  float* ebuf = (float*)alloc((size_t)E_TOT * 4 * 4);
  (void)ws_size; (void)in_sizes; (void)n_in; (void)out_size;

  const dim3 blk(256);
  const int gemmBlocks     = N_NODES / 16;          // 3125 (exact)
  const int nodeWaveBlocks = (N_NODES + 7) / 8;
  const int edgeBlocks     = (E_TOT + 255) / 256;
  const int edgeWaveBlocks = (E_TOT + 7) / 8;
  const int fullBlocks     = (N_NODES * DIM + 255) / 256;

  convert_f16_kernel<<<fullBlocks, blk, 0, stream>>>(x, inbuf);

  for (int l = 0; l < 3; ++l) {
    const int H = (l == 2) ? 1 : 4;
    transpose_w_kernel<<<DIM * DIM / 256, blk, 0, stream>>>(Wl[l], Wt);
    gemm128_kernel<<<gemmBlocks, blk, 0, stream>>>(inbuf, Wt, hfeat);
    node_prep_kernel<<<nodeWaveBlocks, blk, 0, stream>>>(
        hfeat, asl[l], adl[l], als, ald, mb, zb, acc, H);
    if (H == 4) {
      edge_logit_max_kernel<4><<<edgeBlocks, blk, 0, stream>>>(srcI, dstI, als,
                                                               ald, ebuf, mb);
      edge_sum_kernel<4><<<edgeBlocks, blk, 0, stream>>>(dstI, ebuf, mb, zb);
      edge_aggr_kernel<4><<<edgeWaveBlocks, blk, 0, stream>>>(
          srcI, dstI, ebuf, mb, zb, hfeat, acc);
    } else {
      edge_logit_max_kernel<1><<<edgeBlocks, blk, 0, stream>>>(srcI, dstI, als,
                                                               ald, ebuf, mb);
      edge_sum_kernel<1><<<edgeBlocks, blk, 0, stream>>>(dstI, ebuf, mb, zb);
      edge_aggr_kernel<1><<<edgeWaveBlocks, blk, 0, stream>>>(
          srcI, dstI, ebuf, mb, zb, hfeat, acc);
    }
    if (l < 2) {
      relu_bias_kernel<<<fullBlocks, blk, 0, stream>>>(acc, bl[l], inbuf);
    }
  }
  ln_kernel<<<nodeWaveBlocks, blk, 0, stream>>>(acc, bl[2], gam, bet,
                                                (float*)d_out);
}
